// NonLocalAttention_3075196584446
// MI455X (gfx1250) — matrix operand order repaired and checked by reading, not compiled, not run on hardware
//
#include <hip/hip_runtime.h>
#include <hip/hip_bf16.h>

// ---------------------------------------------------------------------------
// NonLocalAttention (channel attention) for MI455X / gfx1250, wave32 + WMMA.
//   pass 0: W -> bf16, x -> xT (B,S,C) bf16   (one conversion, halves traffic)
//   q/k/v = BN(W @ x)        : bf16 WMMA GEMM, async-LDS double-buffered
//   attn  = softmax(q k^T)   : split-K WMMA GEMM + f32 atomics, f32 softmax
//   out   = attn v + x       : bf16 WMMA GEMM, f32 out
// ---------------------------------------------------------------------------

typedef __bf16 bf16_t;
typedef bf16_t bf16x16 __attribute__((ext_vector_type(16)));
typedef float  floatx8 __attribute__((ext_vector_type(8)));
typedef int    intx4   __attribute__((ext_vector_type(4)));

union Frag { bf16x16 v; unsigned int u[8]; };

#define C_DIM 256
#define S_DIM 16384
#define EPS_BN 1e-5f
#define BKP 40   // padded LDS K-stride (bf16): 80B rows -> 16B-aligned b128 chunks,
                 // dword stride 20 -> conflict-free WMMA fragment loads

#if defined(__HIP_DEVICE_COMPILE__) && __has_builtin(__builtin_amdgcn_global_load_async_to_lds_b128)
#define USE_ASYNC 1
#else
#define USE_ASYNC 0
#endif

__device__ __forceinline__ unsigned short f2bf(float f) {
    unsigned int u = __float_as_uint(f);
    return (unsigned short)((u + 0x7FFFu + ((u >> 16) & 1u)) >> 16);
}

// 16-byte global -> LDS copy: async DMA path on gfx1250, plain copy fallback.
__device__ __forceinline__ void cp16(const unsigned short* g, unsigned short* l) {
#if USE_ASYNC
    __builtin_amdgcn_global_load_async_to_lds_b128(
        (__attribute__((address_space(1))) intx4*)g,
        (__attribute__((address_space(3))) intx4*)l, 0, 0);
#else
    *(uint4*)l = *(const uint4*)g;
#endif
}

__device__ __forceinline__ void wait_async() {
#if USE_ASYNC
#if __has_builtin(__builtin_amdgcn_s_wait_asynccnt)
    __builtin_amdgcn_s_wait_asynccnt(0);
#else
    asm volatile("s_wait_asynccnt 0x0" ::: "memory");
#endif
#endif
}

__device__ __forceinline__ floatx8 wmma_bf16(const Frag& a, const Frag& b, floatx8 c) {
    return __builtin_amdgcn_wmma_f32_16x16x32_bf16(false, a.v, false, b.v,
                                                   (short)0, c, false, false);
}

// ---------------------------------------------------------------------------
// Pass 0a: Wq/Wk/Wv (256x256 f32) -> bf16.  grid=(192), block=256.
// ---------------------------------------------------------------------------
__global__ __launch_bounds__(256)
void convert_w_bf16(const float* __restrict__ Wq, const float* __restrict__ Wk,
                    const float* __restrict__ Wv, unsigned short* __restrict__ wb) {
    int g   = blockIdx.x * 256 + threadIdx.x;   // float4 chunk id, 3*16384 total
    int m   = g >> 14;
    int off = (g & 16383) * 4;
    const float* src = (m == 0) ? Wq : (m == 1) ? Wk : Wv;
    float4 w = *(const float4*)&src[off];
    unsigned short* d = &wb[m * 65536 + off];
    d[0] = f2bf(w.x); d[1] = f2bf(w.y); d[2] = f2bf(w.z); d[3] = f2bf(w.w);
}

// ---------------------------------------------------------------------------
// Pass 0b: x (B,C,S) f32 -> xT (B,S,C) bf16, 64x64 LDS-tiled transpose.
// grid=(S/64, C/64, B), block=256.
// ---------------------------------------------------------------------------
__global__ __launch_bounds__(256)
void xpose_bf16(const float* __restrict__ x, unsigned short* __restrict__ xT) {
    const int b  = blockIdx.z;
    const int c0 = blockIdx.y * 64;
    const int s0 = blockIdx.x * 64;
    const int tid = threadIdx.x;
    __shared__ __align__(16) unsigned short t[64 * 72];
#pragma unroll
    for (int i = 0; i < 4; ++i) {
        int idx = tid + i * 256;                // 1024 float4 chunks
        int c   = idx >> 4;
        int s4  = (idx & 15) * 4;
        float4 v = *(const float4*)&x[((size_t)b * C_DIM + c0 + c) * S_DIM + s0 + s4];
        t[(s4 + 0) * 72 + c] = f2bf(v.x);
        t[(s4 + 1) * 72 + c] = f2bf(v.y);
        t[(s4 + 2) * 72 + c] = f2bf(v.z);
        t[(s4 + 3) * 72 + c] = f2bf(v.w);
    }
    __syncthreads();
#pragma unroll
    for (int i = 0; i < 2; ++i) {
        int idx = tid + i * 256;                // 512 uint4 chunks
        int s   = idx >> 3;
        int c8  = (idx & 7) * 8;
        *(uint4*)&xT[((size_t)b * S_DIM + s0 + s) * C_DIM + c0 + c8] =
            *(uint4*)&t[s * 72 + c8];
    }
}

// ---------------------------------------------------------------------------
// Kernel 1: y = BN(Wb @ x) from bf16 operands (A=Wb (C,C), B=xT tile [s][c]).
// Block tile 64(M) x 128(N), K-step 32, double-buffered async LDS staging.
// grid=(S/128, C/64, B), block=256 (8 waves). Wave w: M-tile w&3, N-tiles (w>>2)*4..+3.
// ---------------------------------------------------------------------------
__global__ __launch_bounds__(256)
void qkv_gemm_bn(const unsigned short* __restrict__ xT,  // (B,S,C) bf16
                 const unsigned short* __restrict__ Wb,  // (C,C) bf16
                 const float* __restrict__ gma, const float* __restrict__ bta,
                 const float* __restrict__ mu,  const float* __restrict__ var,
                 unsigned short* __restrict__ y /* (B,C,S) bf16 */) {
    const int tid  = threadIdx.x;
    const int b    = blockIdx.z;
    const int gm0  = blockIdx.y * 64;
    const int gs0  = blockIdx.x * 128;
    const int wave = tid >> 5, lane = tid & 31;
    const int wm = wave & 3, wn2 = wave >> 2;
    const int l15 = lane & 15, lhalf = lane >> 4;

    __shared__ __align__(16) unsigned short sA[2][64 * BKP];   // W tile  [m][c]
    __shared__ __align__(16) unsigned short sB[2][128 * BKP];  // xT tile [s][c]

    const floatx8 zf = {0.f, 0.f, 0.f, 0.f, 0.f, 0.f, 0.f, 0.f};
    floatx8 acc[4];
#pragma unroll
    for (int t = 0; t < 4; ++t) acc[t] = zf;

    const int srow = tid >> 1, sc8 = (tid & 1) * 8;
    auto stage = [&](int bufi, int kk) {
        if (tid < 128)
            cp16(&Wb[(size_t)(gm0 + srow) * C_DIM + kk + sc8], &sA[bufi][srow * BKP + sc8]);
        cp16(&xT[((size_t)b * S_DIM + gs0 + srow) * C_DIM + kk + sc8],
             &sB[bufi][srow * BKP + sc8]);
    };

    stage(0, 0);
    int buf = 0;
    for (int kk = 0; kk < C_DIM; kk += 32) {
        wait_async();
        __syncthreads();
        if (kk + 32 < C_DIM) stage(buf ^ 1, kk + 32);
        Frag a;
        {
            int mrow = (wm * 16 + l15) * BKP;
            int kb0  = 16 * lhalf;
#pragma unroll
            for (int j = 0; j < 4; ++j) {
                a.u[j]     = *(const unsigned int*)&sA[buf][mrow + kb0 + 2 * j];
                a.u[j + 4] = *(const unsigned int*)&sA[buf][mrow + kb0 + 8 + 2 * j];
            }
        }
#pragma unroll
        for (int t = 0; t < 4; ++t) {
            Frag bfr;
            int nrow = ((wn2 * 4 + t) * 16 + l15) * BKP;
            int kb0  = 16 * lhalf;
#pragma unroll
            for (int j = 0; j < 8; ++j)
                bfr.u[j] = *(const unsigned int*)&sB[buf][nrow + kb0 + 2 * j];
            acc[t] = wmma_bf16(a, bfr, acc[t]);
        }
        buf ^= 1;
    }
    // epilogue: inference BatchNorm + bf16 store
#pragma unroll
    for (int r = 0; r < 8; ++r) {
        int o = gm0 + wm * 16 + r + 8 * lhalf;
        float inv  = gma[o] * rsqrtf(var[o] + EPS_BN);
        float bias = bta[o] - mu[o] * inv;
#pragma unroll
        for (int t = 0; t < 4; ++t) {
            float val = acc[t][r] * inv + bias;
            y[((size_t)b * C_DIM + o) * S_DIM + gs0 + (wn2 * 4 + t) * 16 + l15] = f2bf(val);
        }
    }
}

// ---------------------------------------------------------------------------
// Kernel 2: logits[b,i,j] += sum_s q[b,i,s]*k[b,j,s] (NT GEMM, split-K=32).
// Block tile 128x128, wave w -> M-tile w, all 8 N-tiles; async double-buffered.
// grid=(32, 4, B), block=256.
// ---------------------------------------------------------------------------
__global__ __launch_bounds__(256)
void attn_logits(const unsigned short* __restrict__ q,
                 const unsigned short* __restrict__ k,
                 float* __restrict__ L) {
    const int tid  = threadIdx.x;
    const int b    = blockIdx.z;
    const int gi0  = (blockIdx.y & 1) * 128;
    const int gj0  = (blockIdx.y >> 1) * 128;
    const int s0   = blockIdx.x * (S_DIM / 32);
    const int wave = tid >> 5, lane = tid & 31;
    const int l15 = lane & 15, lhalf = lane >> 4;

    __shared__ __align__(16) unsigned short sA[2][128 * BKP];  // q tile [i][s]
    __shared__ __align__(16) unsigned short sB[2][128 * BKP];  // k tile [j][s]

    const floatx8 zf = {0.f, 0.f, 0.f, 0.f, 0.f, 0.f, 0.f, 0.f};
    floatx8 acc[8];
#pragma unroll
    for (int t = 0; t < 8; ++t) acc[t] = zf;

    const int srow = tid >> 1, sc8 = (tid & 1) * 8;
    auto stage = [&](int bufi, int sbase) {
        cp16(&q[((size_t)b * C_DIM + gi0 + srow) * S_DIM + sbase + sc8],
             &sA[bufi][srow * BKP + sc8]);
        cp16(&k[((size_t)b * C_DIM + gj0 + srow) * S_DIM + sbase + sc8],
             &sB[bufi][srow * BKP + sc8]);
    };

    stage(0, s0);
    int buf = 0;
    for (int ks = 0; ks < S_DIM / 32; ks += 32) {
        wait_async();
        __syncthreads();
        if (ks + 32 < S_DIM / 32) stage(buf ^ 1, s0 + ks + 32);
        Frag a;
        {
            int mrow = (wave * 16 + l15) * BKP;
            int kb0  = 16 * lhalf;
#pragma unroll
            for (int j = 0; j < 4; ++j) {
                a.u[j]     = *(const unsigned int*)&sA[buf][mrow + kb0 + 2 * j];
                a.u[j + 4] = *(const unsigned int*)&sA[buf][mrow + kb0 + 8 + 2 * j];
            }
        }
#pragma unroll
        for (int nt = 0; nt < 8; ++nt) {
            Frag bfr;
            int nrow = (nt * 16 + l15) * BKP;
            int kb0  = 16 * lhalf;
#pragma unroll
            for (int j = 0; j < 8; ++j)
                bfr.u[j] = *(const unsigned int*)&sB[buf][nrow + kb0 + 2 * j];
            acc[nt] = wmma_bf16(a, bfr, acc[nt]);
        }
        buf ^= 1;
    }
#pragma unroll
    for (int nt = 0; nt < 8; ++nt)
#pragma unroll
        for (int r = 0; r < 8; ++r) {
            int gi = gi0 + wave * 16 + r + 8 * lhalf;
            int gj = gj0 + nt * 16 + l15;
            atomicAdd(&L[((size_t)b * C_DIM + gi) * C_DIM + gj], acc[nt][r]);
        }
}

// ---------------------------------------------------------------------------
// Kernel 3: row softmax over C=256, f32 -> bf16 attn.  grid=(256,B), block=256
// ---------------------------------------------------------------------------
__global__ __launch_bounds__(256)
void softmax_rows(const float* __restrict__ L, unsigned short* __restrict__ A) {
    const int b = blockIdx.y, i = blockIdx.x, j = threadIdx.x;
    __shared__ float red[256];
    float v = L[((size_t)b * C_DIM + i) * C_DIM + j];
    red[j] = v;
    __syncthreads();
    for (int s = 128; s > 0; s >>= 1) {
        if (j < s) red[j] = fmaxf(red[j], red[j + s]);
        __syncthreads();
    }
    float m = red[0];
    __syncthreads();
    float e = __expf(v - m);
    red[j] = e;
    __syncthreads();
    for (int s = 128; s > 0; s >>= 1) {
        if (j < s) red[j] += red[j + s];
        __syncthreads();
    }
    A[((size_t)b * C_DIM + i) * C_DIM + j] = f2bf(e / red[0]);
}

// ---------------------------------------------------------------------------
// Kernel 4: out = attn @ v + x (f32). A=attn bf16 (async staged), B=v bf16
// transposed into LDS. grid=(S/128, C/64, B), block=256.
// ---------------------------------------------------------------------------
__global__ __launch_bounds__(256)
void attn_v_out(const unsigned short* __restrict__ At,
                const unsigned short* __restrict__ v,
                const float* __restrict__ x, float* __restrict__ out) {
    const int tid  = threadIdx.x;
    const int b    = blockIdx.z;
    const int gm0  = blockIdx.y * 64;
    const int gs0  = blockIdx.x * 128;
    const int wave = tid >> 5, lane = tid & 31;
    const int wm = wave & 3, wn2 = wave >> 2;
    const int l15 = lane & 15, lhalf = lane >> 4;

    __shared__ __align__(16) unsigned short sA[64 * BKP];   // attn tile [i][d]
    __shared__ __align__(16) unsigned short sB[128 * BKP];  // v tile [s][d] (transposed)

    const floatx8 zf = {0.f, 0.f, 0.f, 0.f, 0.f, 0.f, 0.f, 0.f};
    floatx8 acc[4];
#pragma unroll
    for (int t = 0; t < 4; ++t) acc[t] = zf;

    for (int kk = 0; kk < C_DIM; kk += 32) {
        __syncthreads();
        if (tid < 128) {   // stage A: 64x32 bf16 pure copy (async on gfx1250)
            int row = tid >> 1, c8 = (tid & 1) * 8;
            cp16(&At[((size_t)b * C_DIM + gm0 + row) * C_DIM + kk + c8],
                 &sA[row * BKP + c8]);
        }
        // stage B: v 32x128 bf16, transpose into [s][d]
#pragma unroll
        for (int i = 0; i < 2; ++i) {
            int idx = tid + i * 256;            // 512 chunks of 8 bf16
            int row = idx >> 4;                 // d offset 0..31
            int sc  = (idx & 15) * 8;
            uint4 d = *(const uint4*)&v[((size_t)b * C_DIM + kk + row) * S_DIM + gs0 + sc];
            const unsigned short* ds = (const unsigned short*)&d;
#pragma unroll
            for (int u = 0; u < 8; ++u) sB[(sc + u) * BKP + row] = ds[u];
        }
        wait_async();
        __syncthreads();
        Frag a;
        {
            int mrow = (wm * 16 + l15) * BKP;
            int kb0  = 16 * lhalf;
#pragma unroll
            for (int j = 0; j < 4; ++j) {
                a.u[j]     = *(const unsigned int*)&sA[mrow + kb0 + 2 * j];
                a.u[j + 4] = *(const unsigned int*)&sA[mrow + kb0 + 8 + 2 * j];
            }
        }
#pragma unroll
        for (int t = 0; t < 4; ++t) {
            Frag bfr;
            int nrow = ((wn2 * 4 + t) * 16 + l15) * BKP;
            int kb0  = 16 * lhalf;
#pragma unroll
            for (int j = 0; j < 8; ++j)
                bfr.u[j] = *(const unsigned int*)&sB[nrow + kb0 + 2 * j];
            acc[t] = wmma_bf16(a, bfr, acc[t]);
        }
    }
#pragma unroll
    for (int r = 0; r < 8; ++r) {
        int o = gm0 + wm * 16 + r + 8 * lhalf;
#pragma unroll
        for (int t = 0; t < 4; ++t) {
            size_t off = ((size_t)b * C_DIM + o) * S_DIM + gs0 + (wn2 * 4 + t) * 16 + l15;
            out[off] = acc[t][r] + x[off];
        }
    }
}

__global__ __launch_bounds__(256)
void zero_f32(float* __restrict__ p, int n) {
    int i = blockIdx.x * 256 + threadIdx.x;
    if (i < n) p[i] = 0.f;
}

// ---------------------------------------------------------------------------
extern "C" void kernel_launch(void* const* d_in, const int* in_sizes, int n_in,
                              void* d_out, int out_size, void* d_ws, size_t ws_size,
                              hipStream_t stream) {
    const float* x  = (const float*)d_in[0];
    const float* Wk = (const float*)d_in[1];
    const float* kg = (const float*)d_in[2];
    const float* kb = (const float*)d_in[3];
    const float* km = (const float*)d_in[4];
    const float* kv = (const float*)d_in[5];
    const float* Wq = (const float*)d_in[6];
    const float* qg = (const float*)d_in[7];
    const float* qb = (const float*)d_in[8];
    const float* qm = (const float*)d_in[9];
    const float* qv = (const float*)d_in[10];
    const float* Wv = (const float*)d_in[11];
    const float* vg = (const float*)d_in[12];
    const float* vb = (const float*)d_in[13];
    const float* vm = (const float*)d_in[14];
    const float* vv = (const float*)d_in[15];

    const size_t QKV_BYTES = (size_t)8 * C_DIM * S_DIM * 2;   // 64 MB (bf16)
    char* ws = (char*)d_ws;
    unsigned short* q_bf   = (unsigned short*)(ws);
    unsigned short* k_bf   = (unsigned short*)(ws + QKV_BYTES);
    unsigned short* v_bf   = (unsigned short*)(ws + 2 * QKV_BYTES);
    unsigned short* xT     = (unsigned short*)(ws + 3 * QKV_BYTES);
    unsigned short* wbf    = (unsigned short*)(ws + 4 * QKV_BYTES);             // 384 KB
    float*          logits = (float*)(ws + 4 * QKV_BYTES + 393216);            // 2 MB
    unsigned short* attn   = (unsigned short*)(ws + 4 * QKV_BYTES + 393216 +
                                               (size_t)8 * C_DIM * C_DIM * 4); // 1 MB

    dim3 blk(256);
    dim3 gemm_grid(S_DIM / 128, C_DIM / 64, 8);

    convert_w_bf16<<<dim3(192), blk, 0, stream>>>(Wq, Wk, Wv, wbf);
    xpose_bf16<<<dim3(S_DIM / 64, C_DIM / 64, 8), blk, 0, stream>>>(x, xT);

    qkv_gemm_bn<<<gemm_grid, blk, 0, stream>>>(xT, wbf,          qg, qb, qm, qv, q_bf);
    qkv_gemm_bn<<<gemm_grid, blk, 0, stream>>>(xT, wbf + 65536,  kg, kb, km, kv, k_bf);
    qkv_gemm_bn<<<gemm_grid, blk, 0, stream>>>(xT, wbf + 131072, vg, vb, vm, vv, v_bf);

    int nL = 8 * C_DIM * C_DIM;
    zero_f32<<<dim3((nL + 255) / 256), blk, 0, stream>>>(logits, nL);

    attn_logits<<<dim3(32, 4, 8), blk, 0, stream>>>(q_bf, k_bf, logits);
    softmax_rows<<<dim3(C_DIM, 8), blk, 0, stream>>>(logits, attn);
    attn_v_out<<<gemm_grid, blk, 0, stream>>>(attn, v_bf, x, (float*)d_out);
}
